// DistNN_88794153877513
// MI455X (gfx1250) — matrix operand order repaired and checked by reading, not compiled, not run on hardware
//
#include <hip/hip_runtime.h>
#include <hip/hip_bf16.h>

typedef __attribute__((ext_vector_type(16))) _Float16 v16h;
typedef __attribute__((ext_vector_type(8)))  _Float16 v8h;
typedef __attribute__((ext_vector_type(8)))  float    v8f;

union V16 { v16h v; v8h h[2]; };

#define NSEG   50000
#define NGRAPH 8192

// ---------------------------------------------------------------------------
// small utility kernels
// ---------------------------------------------------------------------------
__global__ void zero_f32_kernel(float* __restrict__ p, long n) {
  long i = (long)blockIdx.x * blockDim.x + threadIdx.x;
  if (i < n) p[i] = 0.0f;
}

__global__ void count_f32_kernel(const int* __restrict__ idx, float* __restrict__ cnt, long n) {
  long i = (long)blockIdx.x * blockDim.x + threadIdx.x;
  if (i < n) atomicAdd(&cnt[idx[i]], 1.0f);
}

// segment-sum of h[:,0:32] and h[:,32:64] by atom_idx into pa/pe (L2-resident)
__global__ void pool_sum_kernel(const _Float16* __restrict__ h, const int* __restrict__ idx,
                                float* __restrict__ pa, float* __restrict__ pe, long n) {
  long t = (long)blockIdx.x * blockDim.x + threadIdx.x;
  long i = t >> 6; int c = (int)(t & 63);
  if (i >= n) return;
  long s = idx[i];
  float v = (float)h[i * 64 + c];
  if (c < 32) atomicAdd(&pa[s * 32 + c], v);
  else        atomicAdd(&pe[s * 32 + (c - 32)], v);
}

__global__ void pool_div_kernel(float* __restrict__ pa, float* __restrict__ pe,
                                const float* __restrict__ cnt, long nseg) {
  long t = (long)blockIdx.x * blockDim.x + threadIdx.x;
  long s = t >> 5; int c = (int)(t & 31);
  if (s >= nseg) return;
  float d = 1.0f / fmaxf(cnt[s], 1.0f);
  pa[s * 32 + c] *= d;
  pe[s * 32 + c] *= d;
}

// hcat = [ h[:,0:32] | pa[atom_idx] | h[:,32:64] | pe[ele_idx] | xd ]  (f16, 256 cols)
__global__ void hcat_kernel(const _Float16* __restrict__ h, const float* __restrict__ pa,
                            const float* __restrict__ pe, const _Float16* __restrict__ xd,
                            const int* __restrict__ atom_idx, const int* __restrict__ ele_idx,
                            _Float16* __restrict__ hcat, long n) {
  long t = (long)blockIdx.x * blockDim.x + threadIdx.x;
  long i = t >> 8; int c = (int)(t & 255);
  if (i >= n) return;
  _Float16 v;
  if      (c < 32)  v = h[i * 64 + c];
  else if (c < 64)  v = (_Float16)pa[(long)atom_idx[i] * 32 + (c - 32)];
  else if (c < 96)  v = h[i * 64 + 32 + (c - 64)];
  else if (c < 128) v = (_Float16)pe[(long)ele_idx[i] * 32 + (c - 96)];
  else              v = xd[i * 128 + (c - 128)];
  hcat[i * 256 + c] = v;
}

// fold BN params + column stats into per-column scale/shift (population var, eps=1e-5)
__global__ void bn_finalize_kernel(const float* __restrict__ colsum, const float* __restrict__ colsq,
                                   const float* __restrict__ g, const float* __restrict__ beta,
                                   float* __restrict__ scale, float* __restrict__ shift, float invN) {
  int c = threadIdx.x;   // 64 threads
  float m  = colsum[c] * invN;
  float va = colsq[c] * invN - m * m;
  float sc = g[c] * rsqrtf(va + 1e-5f);
  scale[c] = sc;
  shift[c] = beta[c] - m * sc;
}

// out[i, ocol + c] = relu(y[i,c]*scale[c] + shift[c])  (64 cols fixed)
__global__ void bn_relu_kernel(const float* __restrict__ y, const float* __restrict__ scale,
                               const float* __restrict__ shift, _Float16* __restrict__ out,
                               long ldo, long ocol, long n) {
  long t = (long)blockIdx.x * blockDim.x + threadIdx.x;
  long i = t >> 6; int c = (int)(t & 63);
  if (i >= n) return;
  float v = y[i * 64 + c] * scale[c] + shift[c];
  out[i * ldo + ocol + c] = (_Float16)(v > 0.0f ? v : 0.0f);
}

__global__ void graph_pool_kernel(const _Float16* __restrict__ h, const int* __restrict__ gidx,
                                  float* __restrict__ gsum, long n) {
  long t = (long)blockIdx.x * blockDim.x + threadIdx.x;
  long i = t >> 6; int c = (int)(t & 63);
  if (i >= n) return;
  atomicAdd(&gsum[(long)gidx[i] * 64 + c], (float)h[i * 64 + c]);
}

__global__ void fc1_kernel(const float* __restrict__ gsum, const float* __restrict__ gcnt,
                           const float* __restrict__ w, const float* __restrict__ b,
                           float* __restrict__ out, int ngraph) {
  int g = blockIdx.x * blockDim.x + threadIdx.x;
  if (g >= ngraph) return;
  float inv = 1.0f / fmaxf(gcnt[g], 1.0f);
  float s = b[0];
  #pragma unroll
  for (int c = 0; c < 64; ++c) s += gsum[(long)g * 64 + c] * inv * w[c];
  out[g] = fmaxf(s, 0.0f);
}

// ---------------------------------------------------------------------------
// Generic WMMA GEMM:  Y[M,Nout] = X[M,KREAL] @ W[Nout,KREAL]^T + bias
//   MODE 0: out_f16 = relu(Y)  at [m*ldo + ocol + n]          (embeddings)
//   MODE 1: y_f32 = Y, plus per-column sum / sum-of-squares    (pre-BN)
// Compile-time: NCT = Nout/16, KPAD = K rounded to 32, KREAL = true K, AISF32.
// Block: 256 threads = 8 waves; tile = 128 rows x Nout; K-chunk loop fully
// unrolled. Staging is row-per-thread (tid<128 stages one 32-wide row), so
// every bounds mask is compile-time; the only runtime predication is the
// block-uniform `full` branch for the single partial row tile.
// A tile double-buffered in LDS; f16 path uses global_load_async_to_lds_b128
// (ASYNCcnt) overlapped with the WMMA stream.
// ---------------------------------------------------------------------------
template<int NCT, int KPAD, int KREAL, int AISF32, int MODE>
__global__ void wmma_gemm_kernel(const void* __restrict__ Aptr, long lda, long acol,
                                 const float* __restrict__ W, const float* __restrict__ bias,
                                 long M,
                                 _Float16* __restrict__ outF16, long ldo, long ocol,
                                 float* __restrict__ yF32, float* __restrict__ colsum,
                                 float* __restrict__ colsumsq) {
  constexpr int NOUT = NCT * 16;
  extern __shared__ char smem[];
  _Float16* LW  = (_Float16*)smem;              // [NOUT][KPAD] f16 weights
  _Float16* LA0 = LW + NOUT * KPAD;             // [128][32] f16 A chunk, buffer 0
  _Float16* LA1 = LA0 + 128 * 32;               // buffer 1
  float* LSTAT  = (float*)(LA1 + 128 * 32);     // [2*NOUT] block-local BN stats

  const int tid  = threadIdx.x;
  const int lane = tid & 31;
  const int wave = tid >> 5;

  const long row0 = (long)blockIdx.x * 128;
  const long gr   = row0 + tid;                 // staging row (tid < 128 stages)
  const bool rok  = (gr < M);
  const long grc  = rok ? gr : 0;
  const bool full = (row0 + 128 <= M);          // block-uniform

  // stage one 128x32 A chunk (cols kc..kc+31); kc is compile-time (unrolled)
  auto stage = [&](int kc, _Float16* LAbuf) {
    if (tid < 128) {                            // wave-uniform (waves 0-3)
      if (AISF32) {
        if (KPAD == KREAL) {
          if (full) {
            const float2* Af2 = (const float2*)__builtin_assume_aligned(
                (const float*)Aptr + gr * lda + acol + kc, 8);
            #pragma unroll
            for (int v = 0; v < 4; ++v) {
              v8h q = {};
              #pragma unroll
              for (int e2 = 0; e2 < 4; ++e2) {
                float2 w = Af2[v * 4 + e2];
                q[2 * e2]     = (_Float16)w.x;
                q[2 * e2 + 1] = (_Float16)w.y;
              }
              *(v8h*)&LAbuf[tid * 32 + v * 8] = q;
            }
          } else {
            const float* Af = (const float*)Aptr + grc * lda + acol + kc;
            #pragma unroll
            for (int v = 0; v < 4; ++v) {
              v8h q = {};
              #pragma unroll
              for (int e = 0; e < 8; ++e) {
                float w = Af[v * 8 + e];
                q[e] = (_Float16)(rok ? w : 0.0f);
              }
              *(v8h*)&LAbuf[tid * 32 + v * 8] = q;
            }
          }
        } else {
          // K tail present (embed-atom): k is compile-time, guard folds away
          const float* Af = (const float*)Aptr + grc * lda + acol;
          #pragma unroll
          for (int v = 0; v < 4; ++v) {
            v8h q = {};
            #pragma unroll
            for (int e = 0; e < 8; ++e) {
              int k = kc + v * 8 + e;           // compile-time constant
              float w = (k < KREAL) ? Af[k] : 0.0f;
              q[e] = (_Float16)((k < KREAL && rok) ? w : 0.0f);
            }
            *(v8h*)&LAbuf[tid * 32 + v * 8] = q;
          }
        }
      } else {
        // async DMA: 16B global -> LDS per instruction; instruction offset
        // applies to both addresses, so one base covers the whole 64B row.
        unsigned lds0 = (unsigned)(size_t)&LAbuf[tid * 32];
        unsigned long long ga =
            (unsigned long long)((const _Float16*)Aptr + gr * lda + acol + kc);
        if (rok) {
          asm volatile("global_load_async_to_lds_b128 %0, %1, off"
                       :: "v"(lds0), "v"(ga) : "memory");
          asm volatile("global_load_async_to_lds_b128 %0, %1, off offset:16"
                       :: "v"(lds0), "v"(ga) : "memory");
          asm volatile("global_load_async_to_lds_b128 %0, %1, off offset:32"
                       :: "v"(lds0), "v"(ga) : "memory");
          asm volatile("global_load_async_to_lds_b128 %0, %1, off offset:48"
                       :: "v"(lds0), "v"(ga) : "memory");
        } else {
          v8h z = {};
          #pragma unroll
          for (int v = 0; v < 4; ++v) *(v8h*)&LAbuf[tid * 32 + v * 8] = z;
        }
      }
    }
  };

  // weights -> LDS (f32 -> f16), zero-pad K tail (compile-time fold if no tail)
  for (int i = tid; i < NOUT * KPAD; i += 256) {
    int n = i / KPAD, k = i - n * KPAD;
    if (KPAD == KREAL) {
      LW[i] = (_Float16)W[(long)n * KREAL + k];
    } else {
      int kcl = (k < KREAL) ? k : 0;
      float w = W[(long)n * KREAL + kcl];
      LW[i] = (k < KREAL) ? (_Float16)w : (_Float16)0.0f;
    }
  }
  if (MODE == 1) for (int i = tid; i < 2 * NOUT; i += 256) LSTAT[i] = 0.0f;

  v8f zf = {};
  v8f acc[NCT];
  #pragma unroll
  for (int ct = 0; ct < NCT; ++ct) acc[ct] = zf;

  const int row_in = (wave << 4) + (lane & 15);
  const int akb    = (lane < 16) ? 0 : 8;    // A frag K runs {0-7,16-23} / {8-15,24-31}
  const int bkb    = (lane < 16) ? 0 : 8;    // B frag K runs {0-7,16-23} / {8-15,24-31}

  stage(0, LA0);
  if (!AISF32) asm volatile("s_wait_asynccnt 0" ::: "memory");
  __syncthreads();

  #pragma unroll
  for (int kc = 0; kc < KPAD; kc += 32) {
    _Float16* cur = ((kc >> 5) & 1) ? LA1 : LA0;
    _Float16* nxt = ((kc >> 5) & 1) ? LA0 : LA1;
    constexpr int step = 32;
    const bool more = (kc + step < KPAD);      // compile-time per unrolled iter
    if (more) stage(kc + step, nxt);           // overlap DMA of k+1 with WMMA of k

    V16 a;
    a.h[0] = *(const v8h*)&cur[row_in * 32 + akb];
    a.h[1] = *(const v8h*)&cur[row_in * 32 + akb + 16];
    #pragma unroll
    for (int ct = 0; ct < NCT; ++ct) {
      int n = (ct << 4) + (lane & 15);
      const _Float16* wp = &LW[n * KPAD + kc + bkb];
      V16 b;
      b.h[0] = *(const v8h*)wp;
      b.h[1] = *(const v8h*)(wp + 16);
      acc[ct] = __builtin_amdgcn_wmma_f32_16x16x32_f16(
          false, a.v, false, b.v, (short)0, acc[ct], false, false);
    }

    if (more) {
      if (!AISF32) asm volatile("s_wait_asynccnt 0" ::: "memory");
      __syncthreads();
    }
  }

  // epilogue: C layout — VGPR i holds row (i) for lanes 0-15, (8+i) for lanes 16-31
  const int mbase = (wave << 4) + ((lane < 16) ? 0 : 8);
  #pragma unroll
  for (int ct = 0; ct < NCT; ++ct) {
    int col = (ct << 4) + (lane & 15);
    float bv = bias[col];
    if (MODE == 0) {
      if (full) {
        #pragma unroll
        for (int i = 0; i < 8; ++i) {
          float v = acc[ct][i] + bv;
          outF16[(row0 + mbase + i) * ldo + ocol + col] = (_Float16)(v > 0.0f ? v : 0.0f);
        }
      } else {
        #pragma unroll
        for (int i = 0; i < 8; ++i) {
          long m = row0 + mbase + i;
          if (m < M) {
            float v = acc[ct][i] + bv;
            outF16[m * ldo + ocol + col] = (_Float16)(v > 0.0f ? v : 0.0f);
          }
        }
      }
    } else {
      float s = 0.0f, sq = 0.0f;
      if (full) {
        #pragma unroll
        for (int i = 0; i < 8; ++i) {
          float v = acc[ct][i] + bv;
          yF32[(row0 + mbase + i) * NOUT + col] = v;
          s += v; sq += v * v;
        }
      } else {
        #pragma unroll
        for (int i = 0; i < 8; ++i) {
          long m = row0 + mbase + i;
          if (m < M) {
            float v = acc[ct][i] + bv;
            yF32[m * NOUT + col] = v;
            s += v; sq += v * v;
          }
        }
      }
      atomicAdd(&LSTAT[col], s);          // ds_add_f32
      atomicAdd(&LSTAT[NOUT + col], sq);
    }
  }
  if (MODE == 1) {
    __syncthreads();
    for (int c = tid; c < NOUT; c += 256) {
      atomicAdd(&colsum[c],   LSTAT[c]);
      atomicAdd(&colsumsq[c], LSTAT[NOUT + c]);
    }
  }
}

// ---------------------------------------------------------------------------
// host-side orchestration
// ---------------------------------------------------------------------------
struct DLp { const float *w, *b, *g, *beta; };
struct MLp { DLp main, rdf, bdf; };

using GemmFn = void (*)(const void*, long, long, const float*, const float*,
                        long, _Float16*, long, long,
                        float*, float*, float*);

extern "C" void kernel_launch(void* const* d_in, const int* in_sizes, int n_in,
                              void* d_out, int out_size, void* d_ws, size_t ws_size,
                              hipStream_t stream) {
  (void)n_in; (void)out_size; (void)ws_size;
  const float* atom_feat = (const float*)d_in[0];   // (N,180)
  const float* rdf_feat  = (const float*)d_in[1];   // (N,256)
  const float* bdf_feat  = (const float*)d_in[2];   // (N,256)
  const int* atom_idx  = (const int*)d_in[3];
  const int* ele_idx   = (const int*)d_in[4];
  const int* graph_idx = (const int*)d_in[5];
  // d_in[6] = ref_feat (unused by the reference)
  const long N = in_sizes[3];

  int p = 7;
  auto rd = [&]() -> const float* { return (const float*)d_in[p++]; };
  const float* ea_w = rd();   const float* ea_b = rd();
  const float* erdf_w = rd(); const float* erdf_b = rd();
  const float* ebdf_w = rd(); const float* ebdf_b = rd();
  MLp L[3];
  for (int l = 0; l < 3; ++l) {
    L[l].main.w = rd(); L[l].main.b = rd(); L[l].main.g = rd(); L[l].main.beta = rd();
    L[l].rdf.w  = rd(); L[l].rdf.b  = rd(); L[l].rdf.g  = rd(); L[l].rdf.beta  = rd();
    L[l].bdf.w  = rd(); L[l].bdf.b  = rd(); L[l].bdf.g  = rd(); L[l].bdf.beta  = rd();
  }
  const float* fc_w = rd(); const float* fc_b = rd();

  // workspace carve (fp16 activations keep HBM traffic near the roofline minimum)
  char* wsp = (char*)d_ws;
  auto carve = [&](long bytes) -> void* { void* q = wsp; wsp += (bytes + 255) & ~255L; return q; };
  _Float16* h16  = (_Float16*)carve(N * 64 * 2);
  _Float16* xrdf = (_Float16*)carve(N * 128 * 2);
  _Float16* xbdf = (_Float16*)carve(N * 128 * 2);
  _Float16* hcat = (_Float16*)carve(N * 256 * 2);
  _Float16* x12  = (_Float16*)carve(N * 128 * 2);
  float* y    = (float*)carve(N * 64 * 4);
  float* pa   = (float*)carve((long)NSEG * 32 * 4);
  float* pe   = (float*)carve((long)NSEG * 32 * 4);
  float* cnt  = (float*)carve((long)NSEG * 4);
  float* gsum = (float*)carve((long)NGRAPH * 64 * 4);
  float* gcnt = (float*)carve((long)NGRAPH * 4);
  float* colsum = (float*)carve(64 * 4);
  float* colsq  = (float*)carve(64 * 4);
  float* scale  = (float*)carve(64 * 4);
  float* shift  = (float*)carve(64 * 4);

  auto zero = [&](float* ptr, long n) {
    zero_f32_kernel<<<dim3((unsigned)((n + 255) / 256)), dim3(256), 0, stream>>>(ptr, n);
  };
  auto gemm = [&](GemmFn kern, int Kpad, int Nout,
                  const void* A, long lda, long acol,
                  const float* W, const float* B,
                  _Float16* o16, long ldo, long ocol,
                  float* yo, float* cs, float* cq) {
    size_t lds = (size_t)Nout * Kpad * 2 + 2 * 128 * 32 * 2 + (size_t)2 * Nout * 4;
    unsigned grid = (unsigned)((N + 127) / 128);
    kern<<<dim3(grid), dim3(256), lds, stream>>>(
        A, lda, acol, W, B, N, o16, ldo, ocol, yo, cs, cq);
  };

  // segment / graph counts (once per call; buffers re-zeroed for graph replay)
  zero(cnt, NSEG); zero(gcnt, NGRAPH); zero(gsum, (long)NGRAPH * 64);
  {
    unsigned g = (unsigned)((N + 255) / 256);
    count_f32_kernel<<<g, 256, 0, stream>>>(atom_idx,  cnt,  N);
    count_f32_kernel<<<g, 256, 0, stream>>>(graph_idx, gcnt, N);
  }

  // embeddings (relu -> f16): h = [x_atom | x_ele], x_rdf, x_bdf
  gemm(wmma_gemm_kernel<2, 96, 90, 1, 0>,   96,  32,  atom_feat, 180, 0,  ea_w,   ea_b,
       h16,  64,  0,  nullptr, nullptr, nullptr);
  gemm(wmma_gemm_kernel<2, 96, 90, 1, 0>,   96,  32,  atom_feat, 180, 90, ea_w,   ea_b,
       h16,  64,  32, nullptr, nullptr, nullptr);
  gemm(wmma_gemm_kernel<8, 256, 256, 1, 0>, 256, 128, rdf_feat,  256, 0,  erdf_w, erdf_b,
       xrdf, 128, 0,  nullptr, nullptr, nullptr);
  gemm(wmma_gemm_kernel<8, 256, 256, 1, 0>, 256, 128, bdf_feat,  256, 0,  ebdf_w, ebdf_b,
       xbdf, 128, 0,  nullptr, nullptr, nullptr);

  const unsigned g64   = (unsigned)((N * 64  + 255) / 256);
  const unsigned g256b = (unsigned)((N * 256 + 255) / 256);
  const unsigned gseg  = (unsigned)(((long)NSEG * 32 + 255) / 256);
  const float invN = 1.0f / (float)N;

  for (int l = 0; l < 3; ++l) {
    zero(pa, (long)NSEG * 32); zero(pe, (long)NSEG * 32);
    pool_sum_kernel<<<g64, 256, 0, stream>>>(h16, atom_idx, pa, pe, N);
    pool_div_kernel<<<gseg, 256, 0, stream>>>(pa, pe, cnt, NSEG);

    const _Float16* xd[2] = { xrdf, xbdf };
    const DLp* dl[2] = { &L[l].rdf, &L[l].bdf };
    for (int b = 0; b < 2; ++b) {
      hcat_kernel<<<g256b, 256, 0, stream>>>(h16, pa, pe, xd[b], atom_idx, ele_idx, hcat, N);
      zero(colsum, 64); zero(colsq, 64);
      gemm(wmma_gemm_kernel<4, 256, 256, 0, 1>, 256, 64, hcat, 256, 0,
           dl[b]->w, dl[b]->b, nullptr, 0, 0, y, colsum, colsq);
      bn_finalize_kernel<<<1, 64, 0, stream>>>(colsum, colsq, dl[b]->g, dl[b]->beta,
                                               scale, shift, invN);
      bn_relu_kernel<<<g64, 256, 0, stream>>>(y, scale, shift, x12, 128, (long)(b * 64), N);
    }
    zero(colsum, 64); zero(colsq, 64);
    gemm(wmma_gemm_kernel<4, 128, 128, 0, 1>, 128, 64, x12, 128, 0,
         L[l].main.w, L[l].main.b, nullptr, 0, 0, y, colsum, colsq);
    bn_finalize_kernel<<<1, 64, 0, stream>>>(colsum, colsq, L[l].main.g, L[l].main.beta,
                                             scale, shift, invN);
    bn_relu_kernel<<<g64, 256, 0, stream>>>(y, scale, shift, h16, 64, 0, N);
  }

  graph_pool_kernel<<<g64, 256, 0, stream>>>(h16, graph_idx, gsum, N);
  fc1_kernel<<<dim3((NGRAPH + 255) / 256), dim3(256), 0, stream>>>(
      gsum, gcnt, fc_w, fc_b, (float*)d_out, NGRAPH);
}
